// PixArtBlock_7791070675858
// MI455X (gfx1250) — matrix operand order repaired and checked by reading, not compiled, not run on hardware
//
#include <hip/hip_runtime.h>
#include <math.h>

// ---------------------------------------------------------------------------
// PixArt DiT block for MI455X (gfx1250), bf16 WMMA (f32 accumulate).
// All matmuls use V_WMMA_F32_16X16X32_BF16. Workspace use: ~94.5 MB.
//   bigbuf : (B*N) x HID fp32   (qkv out / q + kv out / fc1 hidden)
//   hbuf   : (B*N) x C  fp32    (h / attn_out / cross_out / h2)
//   mods   : B x 6 x C  fp32
// d_out holds the running residual stream x (fully written by proj GEMM
// before it is first read).
// Staging uses clamped (never predicated) b128 loads: out-of-range rows are
// duplicates whose products never reach a stored output element.
// ---------------------------------------------------------------------------

typedef __attribute__((ext_vector_type(16))) __bf16 v16bf;
typedef __attribute__((ext_vector_type(8)))  __bf16 v8bf;
typedef __attribute__((ext_vector_type(4)))  __bf16 v4bf;
typedef __attribute__((ext_vector_type(8)))  float  v8f;
typedef __attribute__((ext_vector_type(4)))  float  v4f;

#define B_   4
#define N_   1024
#define C_   1152
#define H_   16
#define L_   120
#define D_   72
#define DP_  96
#define HID_ 4608

enum { F_GELU = 1, F_RESID = 2, F_GATE = 4 };

__device__ __forceinline__ v16bf pack16(v8bf lo, v8bf hi) {
  v16bf r;
#pragma unroll
  for (int i = 0; i < 8; ++i) { r[i] = lo[i]; r[i + 8] = hi[i]; }
  return r;
}

__device__ __forceinline__ v4bf cvt4(v4f f) {
  v4bf c;
#pragma unroll
  for (int u = 0; u < 4; ++u) c[u] = (__bf16)f[u];
  return c;
}

// ---------------------------------------------------------------------------
// Generic GEMM: C[M,Nn] = op(A[M,K] @ W[K,Nn] + bias) with fused epilogue.
// Block tile 128x128, BK=32, 8 waves (wave32), wave tile 64x32 (4x2 WMMA).
// ---------------------------------------------------------------------------
#define BM 128
#define BN 128
#define BK 32

__global__ __launch_bounds__(256) void gemm_kernel(
    const float* __restrict__ A, const float* __restrict__ W,
    const float* __restrict__ bias, float* __restrict__ Cout,
    const float* __restrict__ resid, const float* __restrict__ gate,
    int M, int Nn, int K, int flags)
{
  __shared__ __attribute__((aligned(16))) __bf16 As[BM][BK + 8];  // row-major
  __shared__ __attribute__((aligned(16))) __bf16 Bs[BN][BK + 8];  // transposed [n][k]

  const int tid  = threadIdx.x;
  const int lane = tid & 31;
  const int wid  = tid >> 5;
  const int wm   = wid >> 2;   // 0..1
  const int wn   = wid & 3;    // 0..3
  const int r    = lane & 15;
  const int hb   = lane >> 4;

  const int rowBase = blockIdx.y * BM;
  const int colBase = blockIdx.x * BN;

  // A staging addresses: thread i of 4 -> row idx4>>3, col4 (idx4&7)*4
  int arow[4], acol[4];
  const float* aptr[4];
#pragma unroll
  for (int i = 0; i < 4; ++i) {
    int idx4 = i * 256 + tid;
    arow[i] = idx4 >> 3;
    acol[i] = (idx4 & 7) << 2;
    int gr = rowBase + arow[i];
    if (gr >= M) gr = M - 1;              // clamp: dup rows never stored
    aptr[i] = A + (size_t)gr * K + acol[i];
  }
  // B staging: one 4x4 block per thread
  const int bk = (tid >> 5) << 2;         // k 0..28
  const int bn = (tid & 31) << 2;         // n 0..124
  const float* bptr = W + (size_t)bk * Nn + colBase + bn;

  v8f acc[4][2] = {};

  for (int k0 = 0; k0 < K; k0 += BK) {
    // batched b128 loads first, then convert+store
    v4f av[4], bv[4];
#pragma unroll
    for (int i = 0; i < 4; ++i) av[i] = *(const v4f*)(aptr[i] + k0);
#pragma unroll
    for (int j = 0; j < 4; ++j) bv[j] = *(const v4f*)(bptr + (size_t)(k0 + j) * Nn);
    if (k0 + BK < K)
      __builtin_prefetch(bptr + (size_t)(k0 + BK) * Nn, 0, 1);   // global_prefetch_b8

#pragma unroll
    for (int i = 0; i < 4; ++i)
      *(v4bf*)&As[arow[i]][acol[i]] = cvt4(av[i]);
#pragma unroll
    for (int u = 0; u < 4; ++u) {         // transpose 4x4 block into Bs[n][k]
      v4bf c;
      c[0] = (__bf16)bv[0][u]; c[1] = (__bf16)bv[1][u];
      c[2] = (__bf16)bv[2][u]; c[3] = (__bf16)bv[3][u];
      *(v4bf*)&Bs[bn + u][bk] = c;
    }
    __syncthreads();

    v16bf bf[2];
#pragma unroll
    for (int nt = 0; nt < 2; ++nt) {
      // B layout (32x16 bf16): half-lane hb holds K {hb*8..+7} then {16+hb*8..+7}
      const __bf16* p = &Bs[wn * 32 + nt * 16 + r][hb * 8];
      bf[nt] = pack16(*(const v8bf*)p, *(const v8bf*)(p + 16));
    }
#pragma unroll
    for (int mt = 0; mt < 4; ++mt) {
      // A layout (16x32 bf16): half-lane hb holds K {hb*8..+7} then {16+hb*8..+7}
      const __bf16* p = &As[wm * 64 + mt * 16 + r][0];
      v16bf af = pack16(*(const v8bf*)(p + hb * 8), *(const v8bf*)(p + 16 + hb * 8));
#pragma unroll
      for (int nt = 0; nt < 2; ++nt)
        acc[mt][nt] = __builtin_amdgcn_wmma_f32_16x16x32_bf16(
            false, af, false, bf[nt], (short)0, acc[mt][nt], false, false);
    }
    __syncthreads();
  }

  // epilogue: C layout: VGPR j -> row j (lanes 0-15) / row j+8 (lanes 16-31)
  float bias2[2], gate2[2];
  const int bidx = rowBase >> 10;         // batch (1024 rows/batch; BM divides it)
#pragma unroll
  for (int nt = 0; nt < 2; ++nt) {
    int col = colBase + wn * 32 + nt * 16 + r;
    bias2[nt] = bias[col];
    gate2[nt] = (flags & F_GATE) ? gate[bidx * 6 * C_ + col] : 1.f;
  }
  const bool fullM = (rowBase + BM) <= M;
#pragma unroll
  for (int mt = 0; mt < 4; ++mt) {
#pragma unroll
    for (int nt = 0; nt < 2; ++nt) {
      int col = colBase + wn * 32 + nt * 16 + r;
#pragma unroll
      for (int j = 0; j < 8; ++j) {
        int row = rowBase + wm * 64 + mt * 16 + j + hb * 8;
        if (fullM || row < M) {
          float v = acc[mt][nt][j] + bias2[nt];
          if (flags & F_GELU) {
            float u = v;
            v = 0.5f * u * (1.f + tanhf(0.7978845608f * (u + 0.044715f * u * u * u)));
          }
          if (flags & F_GATE) v *= gate2[nt];
          if (flags & F_RESID) v += resid[(size_t)row * Nn + col];
          Cout[(size_t)row * Nn + col] = v;
        }
      }
    }
  }
}

// ---------------------------------------------------------------------------
// Flash attention (self & cross). 128 query rows / block, 64-key blocks,
// D=72 padded to 96. Each of 8 waves owns 16 query rows.
// Element (b, n, d) of a tensor lives at ptr[(b*Nrows+n)*stride + extra + h*D + d].
// K/V pad & clamp regions hold garbage by design:
//   - Q is zero-padded in d, so K pad columns multiply by 0.
//   - invalid keys get S=-1e30 before softmax -> P weight exactly 0.
//   - V pad columns (d>=72) are never stored.
// ---------------------------------------------------------------------------
#define AQ  128
#define AKB 64

__global__ __launch_bounds__(256) void attn_kernel(
    const float* __restrict__ qp, int qs, int qe,
    const float* __restrict__ kp, int ks, int ke,
    const float* __restrict__ vp, int vs, int ve,
    float* __restrict__ op, int os,
    int Nq, int Nk, float scale)
{
  __shared__ __attribute__((aligned(16))) __bf16 Ks[AKB][DP_ + 8];      // [key][d]
  __shared__ __attribute__((aligned(16))) __bf16 Vs[DP_][AKB + 8];      // [d][key]
  __shared__ __attribute__((aligned(16))) __bf16 Ps[8][16][AKB + 8];    // per-wave P

  const int tid  = threadIdx.x;
  const int lane = tid & 31;
  const int wid  = tid >> 5;
  const int r    = lane & 15;
  const int hb   = lane >> 4;

  const int b = blockIdx.y >> 4;
  const int h = blockIdx.y & 15;
  const int qrow0 = blockIdx.x * AQ + wid * 16;

  // Q fragments (pre-scaled by D^-0.5, zero-padded past d=71).
  // Runs are 8-aligned: a run starting at d0 is fully valid iff d0 <= 64.
  v16bf qa[3];
  {
    const float* prow = qp + (size_t)(b * Nq + qrow0 + r) * qs + qe + h * D_;
#pragma unroll
    for (int kt = 0; kt < 3; ++kt) {
      v16bf a;
#pragma unroll
      for (int part = 0; part < 2; ++part) {
        int d0 = kt * 32 + part * 16 + hb * 8;
        const float* pp = prow + (d0 <= 64 ? d0 : 64);
        v4f f0 = *(const v4f*)pp;
        v4f f1 = *(const v4f*)(pp + 4);
        float sc = (d0 <= 64) ? scale : 0.f;
#pragma unroll
        for (int u = 0; u < 4; ++u) {
          a[part * 8 + u]     = (__bf16)(f0[u] * sc);
          a[part * 8 + 4 + u] = (__bf16)(f1[u] * sc);
        }
      }
      qa[kt] = a;
    }
  }

  // staging addressing: 4 threads per key, 6 d-groups of 4 floats each
  const int skey = tid >> 2;
  const int sg0  = tid & 3;

  float mrow[8], lrow[8];
#pragma unroll
  for (int j = 0; j < 8; ++j) { mrow[j] = -1e30f; lrow[j] = 0.f; }
  v8f o[6] = {};

  for (int kb = 0; kb < Nk; kb += AKB) {
    // ---- stage K [key][d] and V transposed [d][key], unconditional b128 loads
    int gk = kb + skey; if (gk >= Nk) gk = Nk - 1;            // clamp
    const float* krow = kp + (size_t)(b * Nk + gk) * ks + ke + h * D_;
    const float* vrow = vp + (size_t)(b * Nk + gk) * vs + ve + h * D_;
    v4f kl[6], vl[6];
#pragma unroll
    for (int i = 0; i < 6; ++i) {
      int d4 = (sg0 + 4 * i) << 2;                            // 0..92
      int dc = (d4 <= 68) ? d4 : 68;                          // clamp (pad junk OK)
      kl[i] = *(const v4f*)(krow + dc);
      vl[i] = *(const v4f*)(vrow + dc);
    }
#pragma unroll
    for (int i = 0; i < 6; ++i) {
      int d4 = (sg0 + 4 * i) << 2;
      *(v4bf*)&Ks[skey][d4] = cvt4(kl[i]);
#pragma unroll
      for (int u = 0; u < 4; ++u) Vs[d4 + u][skey] = (__bf16)vl[i][u];
    }
    __syncthreads();

    // ---- S = Q @ K^T   (16 x 64 per wave)
    v8f s[4] = {};
#pragma unroll
    for (int kt = 0; kt < 3; ++kt) {
#pragma unroll
      for (int nt = 0; nt < 4; ++nt) {
        const __bf16* p = &Ks[nt * 16 + r][kt * 32 + hb * 8];
        v16bf kf = pack16(*(const v8bf*)p, *(const v8bf*)(p + 16));
        s[nt] = __builtin_amdgcn_wmma_f32_16x16x32_bf16(
            false, qa[kt], false, kf, (short)0, s[nt], false, false);
      }
    }
    // mask invalid keys (cross-attn: Nk=120)
#pragma unroll
    for (int nt = 0; nt < 4; ++nt) {
      if (kb + nt * 16 + r >= Nk) {
#pragma unroll
        for (int j = 0; j < 8; ++j) s[nt][j] = -1e30f;
      }
    }
    // ---- online softmax: row j lives in VGPR j across the lane's 16-lane half
#pragma unroll
    for (int j = 0; j < 8; ++j) {
      float mx = s[0][j];
#pragma unroll
      for (int nt = 1; nt < 4; ++nt) mx = fmaxf(mx, s[nt][j]);
#pragma unroll
      for (int off = 1; off < 16; off <<= 1) mx = fmaxf(mx, __shfl_xor(mx, off, 32));
      float mnew = fmaxf(mrow[j], mx);
      float corr = __expf(mrow[j] - mnew);
      float psum = 0.f;
#pragma unroll
      for (int nt = 0; nt < 4; ++nt) {
        float pv = __expf(s[nt][j] - mnew);
        s[nt][j] = pv;
        psum += pv;
      }
#pragma unroll
      for (int off = 1; off < 16; off <<= 1) psum += __shfl_xor(psum, off, 32);
      lrow[j] = lrow[j] * corr + psum;
      mrow[j] = mnew;
#pragma unroll
      for (int dt = 0; dt < 6; ++dt) o[dt][j] *= corr;
    }
    // ---- spill P to per-wave LDS slice (C-layout -> A-fragment reshape)
#pragma unroll
    for (int nt = 0; nt < 4; ++nt)
#pragma unroll
      for (int j = 0; j < 8; ++j)
        Ps[wid][j + hb * 8][nt * 16 + r] = (__bf16)s[nt][j];

    // ---- O += P @ V  (K-dim = 64 keys)
#pragma unroll
    for (int kt2 = 0; kt2 < 2; ++kt2) {
      const __bf16* pp = &Ps[wid][r][kt2 * 32 + hb * 8];
      v16bf pf = pack16(*(const v8bf*)pp, *(const v8bf*)(pp + 16));
#pragma unroll
      for (int dt = 0; dt < 6; ++dt) {
        const __bf16* vv = &Vs[dt * 16 + r][kt2 * 32 + hb * 8];
        v16bf vf = pack16(*(const v8bf*)vv, *(const v8bf*)(vv + 16));
        o[dt] = __builtin_amdgcn_wmma_f32_16x16x32_bf16(
            false, pf, false, vf, (short)0, o[dt], false, false);
      }
    }
    __syncthreads();
  }

  // ---- normalize and write (B,N,C) with heads re-interleaved
#pragma unroll
  for (int dt = 0; dt < 6; ++dt) {
    int d = dt * 16 + r;
    if (d < D_) {
#pragma unroll
      for (int j = 0; j < 8; ++j) {
        int row = qrow0 + j + hb * 8;
        op[(size_t)(b * Nq + row) * os + h * D_ + d] = o[dt][j] / lrow[j];
      }
    }
  }
}

// ---------------------------------------------------------------------------
// mods[b,i,c] = sst[i,c] + t[b, i*C+c]
// ---------------------------------------------------------------------------
__global__ void mods_kernel(const float* __restrict__ t,
                            const float* __restrict__ sst,
                            float* __restrict__ mods)
{
  int idx = blockIdx.x * blockDim.x + threadIdx.x;
  if (idx < B_ * 6 * C_) {
    int rem = idx % (6 * C_);
    mods[idx] = sst[rem] + t[idx];
  }
}

// ---------------------------------------------------------------------------
// h[row] = LN(x[row]) * (1 + sc) + sh  (biased variance, eps=1e-6)
// ---------------------------------------------------------------------------
__global__ __launch_bounds__(256) void modln_kernel(
    const float* __restrict__ x, const float* __restrict__ mods,
    float* __restrict__ h, int shIdx, int scIdx)
{
  const int row = blockIdx.x;
  const int b   = row >> 10;   // N = 1024
  const int tid = threadIdx.x;
  const float* xr = x + (size_t)row * C_;
  __shared__ float rs[8], rs2[8];
  float s = 0.f, s2 = 0.f;
  // C = 1152 = 256 threads * 4.5 -> do 4 vector strides of v4f? keep scalar x2
  for (int c = tid; c < C_; c += 256) {
    float v = xr[c];
    s += v; s2 += v * v;
  }
#pragma unroll
  for (int off = 16; off > 0; off >>= 1) {
    s  += __shfl_xor(s,  off, 32);
    s2 += __shfl_xor(s2, off, 32);
  }
  if ((tid & 31) == 0) { rs[tid >> 5] = s; rs2[tid >> 5] = s2; }
  __syncthreads();
  float tot = 0.f, tot2 = 0.f;
#pragma unroll
  for (int i = 0; i < 8; ++i) { tot += rs[i]; tot2 += rs2[i]; }
  const float mean = tot * (1.f / C_);
  const float var  = tot2 * (1.f / C_) - mean * mean;
  const float rstd = rsqrtf(var + 1e-6f);
  const float* sc = mods + (b * 6 + scIdx) * C_;
  const float* sh = mods + (b * 6 + shIdx) * C_;
  for (int c = tid; c < C_; c += 256)
    h[(size_t)row * C_ + c] = (xr[c] - mean) * rstd * (1.f + sc[c]) + sh[c];
}

// ---------------------------------------------------------------------------
extern "C" void kernel_launch(void* const* d_in, const int* in_sizes, int n_in,
                              void* d_out, int out_size, void* d_ws, size_t ws_size,
                              hipStream_t stream) {
  const float* x       = (const float*)d_in[0];
  const float* y       = (const float*)d_in[1];
  const float* t       = (const float*)d_in[2];
  const float* sst     = (const float*)d_in[3];
  const float* qkv_w   = (const float*)d_in[4];
  const float* qkv_b   = (const float*)d_in[5];
  const float* proj_w  = (const float*)d_in[6];
  const float* proj_b  = (const float*)d_in[7];
  const float* q_w     = (const float*)d_in[8];
  const float* q_b     = (const float*)d_in[9];
  const float* kv_w    = (const float*)d_in[10];
  const float* kv_b    = (const float*)d_in[11];
  const float* cproj_w = (const float*)d_in[12];
  const float* cproj_b = (const float*)d_in[13];
  const float* fc1_w   = (const float*)d_in[14];
  const float* fc1_b   = (const float*)d_in[15];
  const float* fc2_w   = (const float*)d_in[16];
  const float* fc2_b   = (const float*)d_in[17];
  float* out = (float*)d_out;

  char* ws = (char*)d_ws;
  const size_t bigElems = (size_t)(B_ * N_) * HID_;        // fc1 hidden
  float* bigbuf = (float*)ws;
  float* hbuf   = (float*)(ws + bigElems * 4);
  float* modsp  = (float*)(ws + bigElems * 4 + (size_t)(B_ * N_) * C_ * 4);
  float* kvbuf  = bigbuf + (size_t)(B_ * N_) * C_;          // after q region

  const float scale = 1.0f / sqrtf((float)D_);
  const int M = B_ * N_;                                    // 4096

  // 1. mods = sst + t
  mods_kernel<<<dim3((B_ * 6 * C_ + 255) / 256), 256, 0, stream>>>(t, sst, modsp);
  // 2. h = modLN(x, sh_msa, sc_msa)
  modln_kernel<<<dim3(M), 256, 0, stream>>>(x, modsp, hbuf, 0, 1);
  // 3. qkv = h @ qkv_w + b            (4096 x 3456 x 1152)
  gemm_kernel<<<dim3(3 * C_ / BN, M / BM), 256, 0, stream>>>(
      hbuf, qkv_w, qkv_b, bigbuf, nullptr, nullptr, M, 3 * C_, C_, 0);
  // 4. self attention -> hbuf
  attn_kernel<<<dim3(N_ / AQ, B_ * H_), 256, 0, stream>>>(
      bigbuf, 3 * C_, 0, bigbuf, 3 * C_, C_, bigbuf, 3 * C_, 2 * C_,
      hbuf, C_, N_, N_, scale);
  // 5. x = x + g_msa * (attn_out @ proj_w + b)   -> d_out
  gemm_kernel<<<dim3(C_ / BN, M / BM), 256, 0, stream>>>(
      hbuf, proj_w, proj_b, out, x, modsp + 2 * C_, M, C_, C_, F_RESID | F_GATE);
  // 6. q = x @ q_w + b                -> bigbuf
  gemm_kernel<<<dim3(C_ / BN, M / BM), 256, 0, stream>>>(
      out, q_w, q_b, bigbuf, nullptr, nullptr, M, C_, C_, 0);
  // 7. kv = y @ kv_w + b              (480 x 2304 x 1152) -> kvbuf
  gemm_kernel<<<dim3(2 * C_ / BN, (B_ * L_ + BM - 1) / BM), 256, 0, stream>>>(
      y, kv_w, kv_b, kvbuf, nullptr, nullptr, B_ * L_, 2 * C_, C_, 0);
  // 8. cross attention -> hbuf
  attn_kernel<<<dim3(N_ / AQ, B_ * H_), 256, 0, stream>>>(
      bigbuf, C_, 0, kvbuf, 2 * C_, 0, kvbuf, 2 * C_, C_,
      hbuf, C_, N_, L_, scale);
  // 9. x = x + cross_out @ cproj_w + b
  gemm_kernel<<<dim3(C_ / BN, M / BM), 256, 0, stream>>>(
      hbuf, cproj_w, cproj_b, out, out, nullptr, M, C_, C_, F_RESID);
  // 10. h2 = modLN(x, sh_mlp, sc_mlp)
  modln_kernel<<<dim3(M), 256, 0, stream>>>(out, modsp, hbuf, 3, 4);
  // 11. hid = gelu(h2 @ fc1_w + b)    (4096 x 4608 x 1152)
  gemm_kernel<<<dim3(HID_ / BN, M / BM), 256, 0, stream>>>(
      hbuf, fc1_w, fc1_b, bigbuf, nullptr, nullptr, M, HID_, C_, F_GELU);
  // 12. x = x + g_mlp * (hid @ fc2_w + b)   (4096 x 1152 x 4608)
  gemm_kernel<<<dim3(C_ / BN, M / BM), 256, 0, stream>>>(
      bigbuf, fc2_w, fc2_b, out, out, modsp + 5 * C_, M, C_, HID_, F_RESID | F_GATE);
}
